// SSNModel_9646496546835
// MI455X (gfx1250) — matrix operand order repaired and checked by reading, not compiled, not run on hardware
//
#include <hip/hip_runtime.h>
#include <hip/hip_bf16.h>
#include <math.h>

typedef _Float16 v16h __attribute__((ext_vector_type(16)));
typedef _Float16 v8h  __attribute__((ext_vector_type(8)));
typedef float    v8f  __attribute__((ext_vector_type(8)));

// ---------------------------------------------------------------------------
// Implicit-GEMM conv 3x3 SAME via V_WMMA_F32_16X16X32_F16.
// act:  f16 HWC   [H][W][Ci]          (Ci multiple of 32)
// wt:   f16       [9][CoP][Ci]        (tap-major, co, then contiguous ci)
// out:  f32 HWC   [H*W][CoP]
// One wave computes 16 pixels x (NT*16) couts: A fragment loaded once per
// K-chunk and reused across NT WMMAs. Block = 8 waves = 128 pixels.
// All shape params are compile-time so div/mod lower to shifts and the
// tap x K-chunk loop fully unrolls into a dense WMMA stream.
// ---------------------------------------------------------------------------
template <int H, int W, int Ci, int CoP, int NT>
__global__ void __launch_bounds__(256) conv_wmma_k(
    const _Float16* __restrict__ act, const _Float16* __restrict__ wt,
    float* __restrict__ out) {
  const int tid  = threadIdx.x;
  const int wave = tid >> 5;
  const int lane = tid & 31;
  const int pixBase = blockIdx.x * 128 + wave * 16;

  const int m = pixBase + (lane & 15);      // A-matrix row (pixel) for this lane
  const int h = m / W, w = m % W;           // W is a power-of-two constant
  const bool hi  = lane >= 16;
  const int  kbA = hi ? 16 : 0;             // A: lanes16-31 hold K=16..31
  const int  co0 = lane & 15;               // B column within a 16-co tile
  const _Float16* bbase = wt + (size_t)co0 * Ci + (hi ? 16 : 0); // K=16..31 half

  v8f acc[NT];
#pragma unroll
  for (int nt = 0; nt < NT; ++nt) acc[nt] = v8f{};

#pragma unroll
  for (int t = 0; t < 9; ++t) {
    const int dy = t / 3 - 1, dx = t % 3 - 1;
    const int y = h + dy, x = w + dx;
    const bool inb = ((unsigned)y < (unsigned)H) && ((unsigned)x < (unsigned)W);
    const _Float16* ap = act + (size_t)(y * W + x) * Ci + kbA;
    const _Float16* bp = bbase + (size_t)t * CoP * Ci;
#pragma unroll
    for (int cc = 0; cc < Ci; cc += 32) {
      v8h a0 = {}, a1 = {};
      if (inb) {
        a0 = *(const v8h*)(ap + cc);        // K = kbA+0 .. kbA+7
        a1 = *(const v8h*)(ap + cc + 8);    // K = kbA+8 .. kbA+15
      }
      v16h A = __builtin_shufflevector(a0, a1,
                                       0,1,2,3,4,5,6,7,8,9,10,11,12,13,14,15);
#pragma unroll
      for (int nt = 0; nt < NT; ++nt) {
        v16h B = *(const v16h*)(bp + (size_t)nt * 16 * Ci + cc);
        acc[nt] = __builtin_amdgcn_wmma_f32_16x16x32_f16(
            false, A, false, B, (short)0, acc[nt], false, false);
      }
    }
  }
  // C/D layout: VGPR r -> M = r (+8 for lanes 16-31); N = lane&15
#pragma unroll
  for (int nt = 0; nt < NT; ++nt) {
    float* op = out + (size_t)(pixBase + (hi ? 8 : 0)) * CoP + nt * 16 + co0;
#pragma unroll
    for (int r = 0; r < 8; ++r) op[(size_t)r * CoP] = acc[nt][r];
  }
}

// ---------------------------------------------------------------------------
// Weight repack: fp32 OIHW [Co][Ci][3][3] -> f16 [9][CoP][CiP], zero-padded.
// ---------------------------------------------------------------------------
__global__ void cvtw_k(const float* __restrict__ src, _Float16* __restrict__ dst,
                       int Co, int Ci, int CoP, int CiP, int total) {
  int i = blockIdx.x * blockDim.x + threadIdx.x;
  if (i >= total) return;
  int ci = i % CiP;
  int t2 = i / CiP;
  int co = t2 % CoP;
  int t  = t2 / CoP;
  float v = (co < Co && ci < Ci) ? src[((size_t)co * Ci + ci) * 9 + t] : 0.f;
  dst[i] = (_Float16)v;
}

__global__ void fill_k(float* __restrict__ p, float v, int n) {
  int i = blockIdx.x * blockDim.x + threadIdx.x;
  if (i < n) p[i] = v;
}

// conv1: 5->64 direct (tiny FLOPs). x: fp32 CHW (5,256,256); out fp32 HWC.
__global__ void conv1_k(const float* __restrict__ x, const float* __restrict__ w,
                        float* __restrict__ out) {
  int p  = blockIdx.x;          // 65536 pixels
  int co = threadIdx.x;         // 64 couts
  int h = p >> 8, wq = p & 255;
  float s = 0.f;
#pragma unroll
  for (int ci = 0; ci < 5; ++ci)
#pragma unroll
    for (int t = 0; t < 9; ++t) {
      int y = h + t / 3 - 1, xx = wq + t % 3 - 1;
      if ((unsigned)y < 256u && (unsigned)xx < 256u)
        s += x[ci * 65536 + y * 256 + xx] * w[(co * 5 + ci) * 9 + t];
    }
  out[(size_t)p * 64 + co] = s;
}

// BN statistics: per-channel sum & sumsq (C=64). stats[0..63]=sum,[64..127]=sumsq
__global__ void bnstat_k(const float* __restrict__ act, float* __restrict__ stats,
                         int Npix) {
  __shared__ float ssum[256], ssq[256];
  int tid = threadIdx.x;
  int c = tid & 63, sub = tid >> 6;
  float s = 0.f, q = 0.f;
  for (int p = blockIdx.x * 4 + sub; p < Npix; p += gridDim.x * 4) {
    float v = act[(size_t)p * 64 + c];
    s += v; q += v * v;
  }
  ssum[tid] = s; ssq[tid] = q;
  __syncthreads();
  if (sub == 0) {
    for (int k = 1; k < 4; ++k) { s += ssum[c + k * 64]; q += ssq[c + k * 64]; }
    atomicAdd(&stats[c], s);
    atomicAdd(&stats[64 + c], q);
  }
}

__global__ void bnfin_k(const float* __restrict__ stats, const float* __restrict__ g,
                        const float* __restrict__ b, float* __restrict__ sc, int Npix) {
  int c = threadIdx.x;   // 64
  float inv  = 1.f / (float)Npix;
  float mean = stats[c] * inv;
  float var  = stats[64 + c] * inv - mean * mean;
  float s    = g[c] * rsqrtf(var + 1e-5f);
  sc[c]      = s;
  sc[64 + c] = b[c] - mean * s;
}

__global__ void bnrelu_k(const float* __restrict__ act, const float* __restrict__ sc,
                         _Float16* __restrict__ out, int n) {
  int i = blockIdx.x * blockDim.x + threadIdx.x;
  if (i >= n) return;
  int c = i & 63;
  float v = fmaf(act[i], sc[c], sc[64 + c]);
  out[i] = (_Float16)fmaxf(v, 0.f);
}

// 3x3/2 maxpool, pad 1, f16 HWC (C=64)
__global__ void pool_k(const _Float16* __restrict__ in, _Float16* __restrict__ out,
                       int Hi, int Wi) {
  int Ho = Hi >> 1, Wo = Wi >> 1;
  int i = blockIdx.x * blockDim.x + threadIdx.x;
  if (i >= Ho * Wo * 64) return;
  int c = i & 63, p = i >> 6;
  int ow = p % Wo, oh = p / Wo;
  float m = -1e30f;
#pragma unroll
  for (int dy = 0; dy < 3; ++dy)
#pragma unroll
    for (int dx = 0; dx < 3; ++dx) {
      int y = oh * 2 - 1 + dy, x = ow * 2 - 1 + dx;
      if ((unsigned)y < (unsigned)Hi && (unsigned)x < (unsigned)Wi)
        m = fmaxf(m, (float)in[((size_t)(y * Wi + x)) * 64 + c]);
    }
  out[((size_t)(oh * Wo + ow)) * 64 + c] = (_Float16)m;
}

__device__ __forceinline__ float bilin64(const _Float16* __restrict__ s, int I,
                                         int c, int h, int w) {
  float scale = (float)I * (1.f / 256.f);     // half-pixel convention
  float sy = (h + 0.5f) * scale - 0.5f;
  float sx = (w + 0.5f) * scale - 0.5f;
  int y0 = (int)floorf(sy), x0 = (int)floorf(sx);
  float fy = sy - y0, fx = sx - x0;
  int y1 = min(y0 + 1, I - 1), x1 = min(x0 + 1, I - 1);
  y0 = max(y0, 0); x0 = max(x0, 0);
  float v00 = (float)s[((size_t)(y0 * I + x0)) * 64 + c];
  float v01 = (float)s[((size_t)(y0 * I + x1)) * 64 + c];
  float v10 = (float)s[((size_t)(y1 * I + x0)) * 64 + c];
  float v11 = (float)s[((size_t)(y1 * I + x1)) * 64 + c];
  return v00 * (1 - fy) * (1 - fx) + v01 * (1 - fy) * fx +
         v10 * fy * (1 - fx) + v11 * fy * fx;
}

// cat = [x(5) | s1(64) | up(s2)(64) | up(s3)(64) | zero-pad to 224], f16 HWC
__global__ void upcat_k(const float* __restrict__ x, const _Float16* __restrict__ s1,
                        const _Float16* __restrict__ s2, const _Float16* __restrict__ s3,
                        _Float16* __restrict__ cat) {
  int i = blockIdx.x * blockDim.x + threadIdx.x;
  if (i >= 65536 * 224) return;
  int c = i % 224, p = i / 224;
  int h = p >> 8, w = p & 255;
  float v;
  if      (c < 5)   v = x[c * 65536 + p];
  else if (c < 69)  v = (float)s1[(size_t)p * 64 + (c - 5)];
  else if (c < 133) v = bilin64(s2, 128, c - 69, h, w);
  else if (c < 197) v = bilin64(s3, 64, c - 133, h, w);
  else              v = 0.f;
  cat[(size_t)p * 224 + c] = (_Float16)v;
}

// pf = concat(conv7+ob (15), x (5)); write CHW output + HWC working copy
__global__ void pfb_k(const float* __restrict__ c7, const float* __restrict__ ob,
                      const float* __restrict__ x, float* __restrict__ pf_hwc,
                      float* __restrict__ out_pf) {
  int i = blockIdx.x * blockDim.x + threadIdx.x;
  if (i >= 65536 * 20) return;
  int c = i % 20, p = i / 20;
  float v = (c < 15) ? c7[(size_t)p * 16 + c] + ob[c] : x[(c - 15) * 65536 + p];
  pf_hwc[(size_t)p * 20 + c] = v;
  out_pf[c * 65536 + p] = v;
}

// spix init: mean over each 16x16 cell. spix layout [256][20]
__global__ void spinit_k(const float* __restrict__ pf, float* __restrict__ spix) {
  __shared__ float acc[20];
  int s = blockIdx.x, tid = threadIdx.x;
  if (tid < 20) acc[tid] = 0.f;
  __syncthreads();
  int bh = s >> 4, bw = s & 15;
  int p = (bh * 16 + (tid >> 4)) * 256 + bw * 16 + (tid & 15);
#pragma unroll
  for (int c = 0; c < 20; ++c) atomicAdd(&acc[c], pf[(size_t)p * 20 + c]);
  __syncthreads();
  if (tid < 20) spix[s * 20 + tid] = acc[tid] * (1.f / 256.f);
}

// One SSN assignment pass over one image row (block = row h, 256 threads = w).
// accum layout [256][21] (20 weighted-feature sums + weight sum).
__global__ void __launch_bounds__(256) assign_k(
    const float* __restrict__ pf, const float* __restrict__ spix,
    float* __restrict__ accum, float* __restrict__ aff_out) {
  __shared__ float sp[3 * 16 * 20];   // spix rows rr-1..rr+1 (clamped)
  __shared__ float ac[48 * 21];       // per-block accumulator over 48 cells
  int h = blockIdx.x, tid = threadIdx.x;
  int rr = h >> 4;
  for (int i = tid; i < 960; i += 256) {
    int c = i % 20, cell = i / 20;
    int row = rr - 1 + cell / 16;
    row = row < 0 ? 0 : (row > 15 ? 15 : row);
    sp[i] = spix[(row * 16 + (cell % 16)) * 20 + c];
  }
  for (int i = tid; i < 1008; i += 256) ac[i] = 0.f;
  __syncthreads();

  int w = tid, p = h * 256 + w, cc = w >> 4;
  float f[20];
#pragma unroll
  for (int c = 0; c < 20; ++c) f[c] = pf[(size_t)p * 20 + c];

  float dist[9]; bool val[9];
  float dmin = 1e30f;
#pragma unroll
  for (int k = 0; k < 9; ++k) {
    int dr = k / 3 - 1, dc = k % 3 - 1;
    int cr = rr + dr, cd = cc + dc;
    bool v = (unsigned)cr < 16u && (unsigned)cd < 16u;
    val[k] = v;
    float d = 1e30f;
    if (v) {
      d = 0.f;
      const float* spc = &sp[((dr + 1) * 16 + cd) * 20];
#pragma unroll
      for (int c = 0; c < 20; ++c) { float t = f[c] - spc[c]; d += t * t; }
    }
    dist[k] = d;
    dmin = fminf(dmin, d);
  }
  float e[9], se = 0.f;
#pragma unroll
  for (int k = 0; k < 9; ++k) {
    float ek = val[k] ? __expf(dmin - dist[k]) : 0.f;
    e[k] = ek; se += ek;
  }
  float inv = 1.f / se;
#pragma unroll
  for (int k = 0; k < 9; ++k) {
    float a = e[k] * inv;
    aff_out[(size_t)p * 9 + k] = a;
    if (val[k]) {
      int dr = k / 3 - 1, dc = k % 3 - 1;
      float* apc = &ac[((dr + 1) * 16 + (cc + dc)) * 21];
#pragma unroll
      for (int c = 0; c < 20; ++c) atomicAdd(&apc[c], a * f[c]);
      atomicAdd(&apc[20], a);
    }
  }
  __syncthreads();
  for (int i = tid; i < 1008; i += 256) {
    int slot = i / 21, c = i % 21;
    int row = rr - 1 + slot / 16;
    if ((unsigned)row < 16u)
      atomicAdd(&accum[(row * 16 + (slot % 16)) * 21 + c], ac[i]);
  }
}

__global__ void spup_k(const float* __restrict__ accum, float* __restrict__ spix) {
  int i = blockIdx.x * blockDim.x + threadIdx.x;
  if (i >= 5120) return;
  int s = i / 20, c = i % 20;
  spix[i] = accum[s * 21 + c] / (accum[s * 21 + 20] + 1e-16f);
}

// Scatter last-iteration aff into dense abs_aff [S][N] + argmax hard labels.
__global__ void scat_k(const float* __restrict__ aff, float* __restrict__ out_aff,
                       float* __restrict__ out_hard) {
  int p = blockIdx.x * blockDim.x + threadIdx.x;
  if (p >= 65536) return;
  int h = p >> 8, w = p & 255;
  int rr = h >> 4, cc = w >> 4;
  float best = -1.f; int bk = 0;
#pragma unroll
  for (int k = 0; k < 9; ++k) {
    float a = aff[(size_t)p * 9 + k];
    if (a > best) { best = a; bk = k; }
    int cr = rr + k / 3 - 1, cd = cc + k % 3 - 1;
    if ((unsigned)cr < 16u && (unsigned)cd < 16u)
      out_aff[(size_t)(cr * 16 + cd) * 65536 + p] = a;
  }
  out_hard[p] = (float)(rr * 16 + cc + (bk / 3 - 1) * 16 + (bk % 3 - 1));
}

__global__ void spout_k(const float* __restrict__ spix, float* __restrict__ out) {
  int i = blockIdx.x * blockDim.x + threadIdx.x;
  if (i >= 5120) return;
  int s = i % 256, c = i / 256;
  out[c * 256 + s] = spix[s * 20 + c];   // (B,C,S) layout
}

// ---------------------------------------------------------------------------
extern "C" void kernel_launch(void* const* d_in, const int* in_sizes, int n_in,
                              void* d_out, int out_size, void* d_ws, size_t ws_size,
                              hipStream_t stream) {
  const float* x   = (const float*)d_in[0];
  const float* cw1 = (const float*)d_in[1];
  const float* cw2 = (const float*)d_in[2];
  const float* cw3 = (const float*)d_in[3];
  const float* cw4 = (const float*)d_in[4];
  const float* cw5 = (const float*)d_in[5];
  const float* cw6 = (const float*)d_in[6];
  const float* cw7 = (const float*)d_in[7];
  const float* ob  = (const float*)d_in[8];
  const float* g[7] = {nullptr, (const float*)d_in[10], (const float*)d_in[12],
                       (const float*)d_in[14], (const float*)d_in[16],
                       (const float*)d_in[18], (const float*)d_in[20]};
  const float* b[7] = {nullptr, (const float*)d_in[11], (const float*)d_in[13],
                       (const float*)d_in[15], (const float*)d_in[17],
                       (const float*)d_in[19], (const float*)d_in[21]};

  // ---- workspace layout ----
  char* ws = (char*)d_ws;
  size_t off = 0;
  auto alloc = [&](size_t bytes) {
    size_t o = off;
    off = (off + bytes + 255) & ~(size_t)255;
    return (void*)(ws + o);
  };
  _Float16* w2 = (_Float16*)alloc(9 * 64 * 64 * 2);
  _Float16* w3 = (_Float16*)alloc(9 * 64 * 64 * 2);
  _Float16* w4 = (_Float16*)alloc(9 * 64 * 64 * 2);
  _Float16* w5 = (_Float16*)alloc(9 * 64 * 64 * 2);
  _Float16* w6 = (_Float16*)alloc(9 * 64 * 64 * 2);
  _Float16* w7 = (_Float16*)alloc(9 * 16 * 224 * 2);
  float*    tconv = (float*)alloc((size_t)65536 * 64 * 4);   // conv out (reused)
  _Float16* s1a = (_Float16*)alloc((size_t)65536 * 64 * 2);
  _Float16* s1  = (_Float16*)alloc((size_t)65536 * 64 * 2);
  _Float16* p1  = (_Float16*)alloc((size_t)16384 * 64 * 2);
  _Float16* s2a = (_Float16*)alloc((size_t)16384 * 64 * 2);
  _Float16* s2  = (_Float16*)alloc((size_t)16384 * 64 * 2);
  _Float16* p2  = (_Float16*)alloc((size_t)4096 * 64 * 2);
  _Float16* s3a = (_Float16*)alloc((size_t)4096 * 64 * 2);
  _Float16* s3  = (_Float16*)alloc((size_t)4096 * 64 * 2);
  _Float16* cat = (_Float16*)alloc((size_t)65536 * 224 * 2);
  float* pf_hwc = (float*)alloc((size_t)65536 * 20 * 4);
  float* aff    = (float*)alloc((size_t)65536 * 9 * 4);
  float* spix   = (float*)alloc(256 * 20 * 4);
  float* accum  = (float*)alloc(256 * 21 * 4);
  float* stats  = (float*)alloc(128 * 4);
  float* bnp    = (float*)alloc(128 * 4);

  float* out_pf   = (float*)d_out;                 // 20*65536
  float* out_aff  = out_pf + 1310720;              // 256*65536
  float* out_hard = out_aff + 16777216;            // 65536
  float* out_spix = out_hard + 65536;              // 20*256

  // ---- weight repack ----
  {
    int tot = 9 * 64 * 64;
    const float* srcs[5] = {cw2, cw3, cw4, cw5, cw6};
    _Float16* dsts[5] = {w2, w3, w4, w5, w6};
    for (int i = 0; i < 5; ++i)
      cvtw_k<<<(tot + 255) / 256, 256, 0, stream>>>(srcs[i], dsts[i], 64, 64, 64, 64, tot);
    int tot7 = 9 * 16 * 224;
    cvtw_k<<<(tot7 + 255) / 256, 256, 0, stream>>>(cw7, w7, 15, 197, 16, 224, tot7);
  }

  auto bn = [&](const float* gg, const float* bb, const float* src, _Float16* dst,
                int Npix) {
    fill_k<<<1, 128, 0, stream>>>(stats, 0.f, 128);
    bnstat_k<<<256, 256, 0, stream>>>(src, stats, Npix);
    bnfin_k<<<1, 64, 0, stream>>>(stats, gg, bb, bnp, Npix);
    int n = Npix * 64;
    bnrelu_k<<<(n + 255) / 256, 256, 0, stream>>>(src, bnp, dst, n);
  };

  // ---- feature extraction ----
  conv1_k<<<65536, 64, 0, stream>>>(x, cw1, tconv);
  bn(g[1], b[1], tconv, s1a, 65536);
  conv_wmma_k<256, 256, 64, 64, 4><<<65536 / 128, 256, 0, stream>>>(s1a, w2, tconv);
  bn(g[2], b[2], tconv, s1, 65536);
  pool_k<<<(16384 * 64 + 255) / 256, 256, 0, stream>>>(s1, p1, 256, 256);
  conv_wmma_k<128, 128, 64, 64, 4><<<16384 / 128, 256, 0, stream>>>(p1, w3, tconv);
  bn(g[3], b[3], tconv, s2a, 16384);
  conv_wmma_k<128, 128, 64, 64, 4><<<16384 / 128, 256, 0, stream>>>(s2a, w4, tconv);
  bn(g[4], b[4], tconv, s2, 16384);
  pool_k<<<(4096 * 64 + 255) / 256, 256, 0, stream>>>(s2, p2, 128, 128);
  conv_wmma_k<64, 64, 64, 64, 4><<<4096 / 128, 256, 0, stream>>>(p2, w5, tconv);
  bn(g[5], b[5], tconv, s3a, 4096);
  conv_wmma_k<64, 64, 64, 64, 4><<<4096 / 128, 256, 0, stream>>>(s3a, w6, tconv);
  bn(g[6], b[6], tconv, s3, 4096);

  {
    int n = 65536 * 224;
    upcat_k<<<(n + 255) / 256, 256, 0, stream>>>(x, s1, s2, s3, cat);
  }
  conv_wmma_k<256, 256, 224, 16, 1><<<65536 / 128, 256, 0, stream>>>(cat, w7, tconv);
  {
    int n = 65536 * 20;
    pfb_k<<<(n + 255) / 256, 256, 0, stream>>>(tconv, ob, x, pf_hwc, out_pf);
  }

  // ---- SSN iterations ----
  spinit_k<<<256, 256, 0, stream>>>(pf_hwc, spix);
  for (int it = 0; it < 10; ++it) {
    fill_k<<<(5376 + 255) / 256, 256, 0, stream>>>(accum, 0.f, 5376);
    assign_k<<<256, 256, 0, stream>>>(pf_hwc, spix, accum, aff);
    spup_k<<<(5120 + 255) / 256, 256, 0, stream>>>(accum, spix);
  }

  // ---- outputs ----
  fill_k<<<(16777216 + 255) / 256, 256, 0, stream>>>(out_aff, 0.f, 16777216);
  scat_k<<<(65536 + 255) / 256, 256, 0, stream>>>(aff, out_aff, out_hard);
  spout_k<<<(5120 + 255) / 256, 256, 0, stream>>>(spix, out_spix);
}
